// Seq2SeqModel_65034394796345
// MI455X (gfx1250) — matrix operand order repaired and checked by reading, not compiled, not run on hardware
//
#include <hip/hip_runtime.h>
#include <hip/hip_bf16.h>

// ---------------------------------------------------------------------------
// Seq2Seq LSTM + attention for MI455X (gfx1250), wave32 + WMMA bf16.
//   B=256, S=512, D=32, H=512, PRED_LEN=24
// Strategy:
//   * Weights converted once to bf16, gate-permuted so each (i,f,g,o) tile
//     quadruple is register-local -> elementwise cell update, no LDS gates.
//   * Encoder: 16 persistent workgroups, 16-batch-row tile each, 512 steps.
//     Gate GEMM processed as 4 fused groups of 4 tiles (32 live acc VGPRs),
//     cross-k-step double-buffered fragments (A + 4xB prefetched while 4
//     WMMAs execute) -> loads stay ahead of v_wmma instead of loadcnt 0 waits.
//     h-state ping-pong LDS buffers -> 2 barriers/step instead of 5.
//   * enc_proj: flat WMMA GEMM (131072 x 512 x 512).
//   * Decoder: persistent, 24 steps; h1@Wdec + both LSTM cells on WMMA.
// ---------------------------------------------------------------------------

typedef __attribute__((ext_vector_type(16))) __bf16 v16bf;
typedef __attribute__((ext_vector_type(8)))  float  v8f;

#define BB 256
#define SS 512
#define DD 32
#define HH 512
#define PL 24
#define MT 16   // batch tile per workgroup

// ---------------- WMMA fragment loaders (ISA 7.12.2 layouts) ----------------
// A (16x32 bf16, row-major source, stride in elements):
//   lane<16 : row m=lane,   elems = K[k0..k0+7], K[k0+16..k0+23]
//   lane>=16: row m=lane-16,elems = K[k0+8..k0+15], K[k0+24..k0+31]
__device__ __forceinline__ v16bf load_frag_a(const __bf16* base, int stride,
                                             int k0, int lane) {
  const int off = (lane & 15) * stride + k0 + ((lane >> 4) << 3);
  const __bf16* p = base + off;
  union { uint4 u[2]; v16bf v; } f;
  f.u[0] = *(const uint4*)(p);
  f.u[1] = *(const uint4*)(p + 16);
  return f.v;
}
// B (32x16 bf16): B[k][n] = W[n0+n][k0+k] (W row-major):
//   lane<16 : col n=lane, K[k0..k0+7], K[k0+16..k0+23];  lane>=16: K[k0+8..k0+15], K[k0+24..k0+31]
__device__ __forceinline__ v16bf load_frag_b_at(const __bf16* base, int off) {
  const __bf16* p = base + off;
  union { uint4 u[2]; v16bf v; } f;
  f.u[0] = *(const uint4*)(p);
  f.u[1] = *(const uint4*)(p + 16);
  return f.v;
}

__device__ __forceinline__ v8f wmma_bf16(v16bf a, v16bf b, v8f c) {
  return __builtin_amdgcn_wmma_f32_16x16x32_bf16(false, a, false, b,
                                                 (short)0, c, false, false);
}

// 4-tile GEMM, cross-k-step double buffered: fragments for k-step ks+1 are
// issued before the 4 WMMAs of k-step ks consume the current set.
template <int KSTEPS>
__device__ __forceinline__ void gemm4_bf16(const __bf16* A, int astride,
                                           const __bf16* W, int wstride,
                                           int n0, int lane, v8f* acc) {
  const int lanebase = ((lane & 15) + n0) * wstride + ((lane >> 4) << 3);
  v16bf af = load_frag_a(A, astride, 0, lane);
  v16bf b0 = load_frag_b_at(W, lanebase);
  v16bf b1 = load_frag_b_at(W, lanebase + 16 * wstride);
  v16bf b2 = load_frag_b_at(W, lanebase + 32 * wstride);
  v16bf b3 = load_frag_b_at(W, lanebase + 48 * wstride);
#pragma unroll 2
  for (int ks = 0; ks < KSTEPS - 1; ++ks) {
    const int kb = lanebase + (ks + 1) * 32;
    const v16bf afn = load_frag_a(A, astride, (ks + 1) * 32, lane);
    const v16bf c0 = load_frag_b_at(W, kb);
    const v16bf c1 = load_frag_b_at(W, kb + 16 * wstride);
    const v16bf c2 = load_frag_b_at(W, kb + 32 * wstride);
    const v16bf c3 = load_frag_b_at(W, kb + 48 * wstride);
    acc[0] = wmma_bf16(af, b0, acc[0]);
    acc[1] = wmma_bf16(af, b1, acc[1]);
    acc[2] = wmma_bf16(af, b2, acc[2]);
    acc[3] = wmma_bf16(af, b3, acc[3]);
    af = afn; b0 = c0; b1 = c1; b2 = c2; b3 = c3;
  }
  acc[0] = wmma_bf16(af, b0, acc[0]);
  acc[1] = wmma_bf16(af, b1, acc[1]);
  acc[2] = wmma_bf16(af, b2, acc[2]);
  acc[3] = wmma_bf16(af, b3, acc[3]);
}

__device__ __forceinline__ float sigm(float x) { return 1.f / (1.f + __expf(-x)); }

// One (i,f,g,o) tile group -> update 8 c values (regs) and scatter 8 h values.
__device__ __forceinline__ void lstm_group(const v8f* acc, float* cr,
                                           int wave, int a, int lane,
                                           __bf16* d1, int s1,
                                           __bf16* d2, int s2,
                                           __bf16* g, size_t gstride) {
  const int laneN = lane & 15;
  const int hi8   = (lane >> 4) << 3;
  const int j = (wave << 6) + (a << 4) + laneN;
  const v8f gi = acc[0], gf = acc[1], gg = acc[2], go = acc[3];
#pragma unroll
  for (int r = 0; r < 8; ++r) {
    const int m = r + hi8;
    const float iv = sigm(gi[r]), fv = sigm(gf[r]);
    const float gv = tanhf(gg[r]), ov = sigm(go[r]);
    const float cv = fv * cr[r] + iv * gv;
    cr[r] = cv;
    const __bf16 hb = (__bf16)(ov * tanhf(cv));
    d1[m * s1 + j] = hb;
    if (d2) d2[m * s2 + j] = hb;
    if (g)  g[(size_t)m * gstride + j] = hb;
  }
}

// ------------------------- weight conversion -------------------------------
// Permutation: n' = chunk*64 + gate*16 + r  <-  orig n = gate*512 + chunk*16 + r
__global__ void __launch_bounds__(256)
k_convert_lstm(const float* __restrict__ W, int Kx, const float* __restrict__ U,
               const float* __restrict__ bi, const float* __restrict__ bh,
               __bf16* __restrict__ out, int Kpad, float* __restrict__ bout) {
  const int np = blockIdx.x;
  const int chunk = np >> 6, g = (np >> 4) & 3, r = np & 15;
  const int n = g * HH + chunk * 16 + r;
  __bf16* orow = out + (size_t)np * Kpad;
  for (int c = threadIdx.x; c < Kpad; c += blockDim.x) {
    float v = 0.f;
    if (c < Kx)            v = W[(size_t)n * Kx + c];
    else if (c < Kx + HH)  v = U[(size_t)n * HH + (c - Kx)];
    orow[c] = (__bf16)v;
  }
  if (threadIdx.x == 0) bout[np] = bi[n] + bh[n];
}

__global__ void __launch_bounds__(256)
k_convert_attn(const float* __restrict__ attnW, __bf16* __restrict__ Wenc,
               __bf16* __restrict__ Wdec) {
  const int k = blockIdx.x;
  for (int h = threadIdx.x; h < HH; h += blockDim.x) {
    Wenc[(size_t)k * HH + h] = (__bf16)attnW[(size_t)k * (2 * HH) + h];
    Wdec[(size_t)k * HH + h] = (__bf16)attnW[(size_t)k * (2 * HH) + HH + h];
  }
}

// ------------------------------ encoder ------------------------------------
// Ping-pong state buffers: step t reads A0[p]/A1[p]; layer0 updates write
// A0[q]+32 and A1[q]+0; layer1 reads A1[q] (fresh h0 | h1 from rotation) and
// writes h1 into A1[p]+512 (dead for reads this step).  2 barriers/step.
__global__ void __launch_bounds__(256)
k_encoder(const float* __restrict__ src,
          const __bf16* __restrict__ Wc0, const float* __restrict__ b0c,
          const __bf16* __restrict__ Wc1, const float* __restrict__ b1c,
          __bf16* __restrict__ encouts, __bf16* __restrict__ h0f,
          __bf16* __restrict__ h1f, float* __restrict__ c0f,
          float* __restrict__ c1f) {
  extern __shared__ char smem[];
  __bf16* A0base = (__bf16*)smem;                          // 2 x (16x544)
  __bf16* A1base = (__bf16*)(smem + 2 * MT * 544 * 2);     // 2 x (16x1024)
  const int A0SZ = MT * 544;
  const int A1SZ = MT * 1024;
  const int tid = threadIdx.x, wave = tid >> 5, lane = tid & 31;
  const int b0 = blockIdx.x * MT;
  const int n0w = wave * 256;
  const int laneN = lane & 15, hi8 = (lane >> 4) << 3;

  float c0r[4][8], c1r[4][8];
#pragma unroll
  for (int a = 0; a < 4; ++a)
#pragma unroll
    for (int r = 0; r < 8; ++r) { c0r[a][r] = 0.f; c1r[a][r] = 0.f; }

  for (int idx = tid; idx < MT * HH; idx += 256) {
    const int m = idx >> 9, j = idx & 511;
    A0base[m * 544 + 32 + j]           = (__bf16)0.f;
    A1base[m * 1024 + j]               = (__bf16)0.f;
    A1base[m * 1024 + 512 + j]         = (__bf16)0.f;
    A1base[A1SZ + m * 1024 + 512 + j]  = (__bf16)0.f;
  }
#pragma unroll
  for (int i = 0; i < 2; ++i) {       // stage x_0
    const int flat = tid * 2 + i;
    const int m = flat >> 5, c = flat & 31;
    A0base[m * 544 + c] = (__bf16)src[((size_t)(b0 + m) * SS) * DD + c];
  }
  __syncthreads();

  for (int t = 0; t < SS; ++t) {
    const int p = t & 1, q = p ^ 1;
    __bf16* A0r = A0base + p * A0SZ;
    __bf16* A0w = A0base + q * A0SZ;
    __bf16* A1r = A1base + p * A1SZ;
    __bf16* A1w = A1base + q * A1SZ;

    if (t + 1 < SS) {                 // stage x_{t+1} into the write buffer
#pragma unroll
      for (int i = 0; i < 2; ++i) {
        const int flat = tid * 2 + i;
        const int m = flat >> 5, c = flat & 31;
        A0w[m * 544 + c] = (__bf16)src[((size_t)(b0 + m) * SS + t + 1) * DD + c];
      }
    }

    // layer 0: four fused (GEMM + cell-update) tile groups
#pragma unroll 1
    for (int a = 0; a < 4; ++a) {
      v8f acc[4];
#pragma unroll
      for (int g4 = 0; g4 < 4; ++g4) {
        const float bv = b0c[n0w + a * 64 + g4 * 16 + laneN];
#pragma unroll
        for (int r = 0; r < 8; ++r) acc[g4][r] = bv;
      }
      gemm4_bf16<17>(A0r, 544, Wc0, 544, n0w + a * 64, lane, acc);
      lstm_group(acc, c0r[a], wave, a, lane, A0w + 32, 544, A1w, 1024,
                 nullptr, 0);
    }
    __syncthreads();   // h0(t) complete before layer1 reads A1w

    // layer 1
#pragma unroll 1
    for (int a = 0; a < 4; ++a) {
      v8f acc[4];
#pragma unroll
      for (int g4 = 0; g4 < 4; ++g4) {
        const float bv = b1c[n0w + a * 64 + g4 * 16 + laneN];
#pragma unroll
        for (int r = 0; r < 8; ++r) acc[g4][r] = bv;
      }
      gemm4_bf16<32>(A1w, 1024, Wc1, 1024, n0w + a * 64, lane, acc);
      lstm_group(acc, c1r[a], wave, a, lane, A1r + 512, 1024, nullptr, 0,
                 encouts + ((size_t)b0 * SS + t) * HH, (size_t)SS * HH);
    }
    __syncthreads();   // end-of-step
  }

  // SS even: final h0 lives in buffer0 of A1, final h1 in buffer1 of A1 +512.
  for (int idx = tid; idx < MT * HH; idx += 256) {
    const int m = idx >> 9, j = idx & 511;
    h0f[(size_t)(b0 + m) * HH + j] = A1base[m * 1024 + j];
    h1f[(size_t)(b0 + m) * HH + j] = A1base[A1SZ + m * 1024 + 512 + j];
  }
#pragma unroll
  for (int a = 0; a < 4; ++a)
#pragma unroll
    for (int r = 0; r < 8; ++r) {
      const int m = r + hi8, j = (wave << 6) + (a << 4) + laneN;
      c0f[(size_t)(b0 + m) * HH + j] = c0r[a][r];
      c1f[(size_t)(b0 + m) * HH + j] = c1r[a][r];
    }
}

// ------------------------------ enc_proj -----------------------------------
__global__ void __launch_bounds__(256)
k_encproj(const __bf16* __restrict__ EO, const __bf16* __restrict__ Wenc,
          const float* __restrict__ attn_b, __bf16* __restrict__ EP) {
  const int tid = threadIdx.x, wave = tid >> 5, lane = tid & 31;
  const size_t m0 = (size_t)blockIdx.x * 32 + (size_t)(wave >> 2) * 16;
  const int n0w = (wave & 3) * 128;
  const int laneN = lane & 15, hi8 = (lane >> 4) << 3;
  const __bf16* Ab = EO + m0 * HH;
#pragma unroll 1
  for (int grp = 0; grp < 2; ++grp) {
    const int n0 = n0w + grp * 64;
    v8f acc[4];
#pragma unroll
    for (int g4 = 0; g4 < 4; ++g4)
#pragma unroll
      for (int r = 0; r < 8; ++r) acc[g4][r] = 0.f;
    gemm4_bf16<16>(Ab, HH, Wenc, HH, n0, lane, acc);
#pragma unroll
    for (int g4 = 0; g4 < 4; ++g4) {
      const float bb = attn_b[n0 + g4 * 16 + laneN];
#pragma unroll
      for (int r = 0; r < 8; ++r) {
        const int m = r + hi8;
        EP[(m0 + m) * HH + n0 + g4 * 16 + laneN] = (__bf16)(acc[g4][r] + bb);
      }
    }
  }
}

// ------------------------------ decoder ------------------------------------
__global__ void __launch_bounds__(256)
k_decoder(const float* __restrict__ src,
          const __bf16* __restrict__ Wd0, const float* __restrict__ bd0c,
          const __bf16* __restrict__ Wd1, const float* __restrict__ bd1c,
          const __bf16* __restrict__ Wdec, const __bf16* __restrict__ EO,
          const __bf16* __restrict__ EP, const __bf16* __restrict__ h0f,
          const __bf16* __restrict__ h1f, const float* __restrict__ c0f,
          const float* __restrict__ c1f, const float* __restrict__ vW,
          const float* __restrict__ fcW, const float* __restrict__ fcb,
          float* __restrict__ out) {
  extern __shared__ char smem[];
  __bf16* A0    = (__bf16*)smem;                          // 16x1056 [din|ctx|h0|pad]
  __bf16* A1    = (__bf16*)(smem + MT * 1056 * 2);        // 16x1024 [h0|h1]
  float*  hproj = (float*)(smem + MT * 1056 * 2 + MT * 1024 * 2);  // 16x512
  float*  scw   = hproj + MT * HH;                        // 16x512
  float*  vv    = scw + MT * HH;                          // 512
  float*  fcw   = vv + HH;                                // 512
  float*  dinS  = fcw + HH;                               // 16
  float*  histS = dinS + MT;                              // 16

  const int tid = threadIdx.x, wave = tid >> 5, lane = tid & 31;
  const int b0 = blockIdx.x * MT;
  const int n0w = wave * 256;
  const int laneN = lane & 15, hi8 = (lane >> 4) << 3;

  if (tid < MT) {
    histS[tid] = src[((size_t)(b0 + tid) * SS + (SS - 1)) * DD + (DD - 15)];
    dinS[tid]  = src[((size_t)(b0 + tid) * SS + (SS - 1)) * DD + (DD - 1)];
  }
  for (int h = tid; h < HH; h += 256) { vv[h] = vW[h]; fcw[h] = fcW[h]; }
  for (int idx = tid; idx < MT * HH; idx += 256) {
    const int m = idx >> 9, j = idx & 511;
    const __bf16 h0v = h0f[(size_t)(b0 + m) * HH + j];
    A1[m * 1024 + j]        = h0v;
    A0[m * 1056 + 513 + j]  = h0v;
    A1[m * 1024 + 512 + j]  = h1f[(size_t)(b0 + m) * HH + j];
  }
  for (int idx = tid; idx < MT * 31; idx += 256) {  // pad cols 1025..1055
    const int m = idx / 31, c = 1025 + idx % 31;
    A0[m * 1056 + c] = (__bf16)0.f;
  }
  float c0r[4][8], c1r[4][8];
#pragma unroll
  for (int a = 0; a < 4; ++a)
#pragma unroll
    for (int r = 0; r < 8; ++r) {
      const int m = r + hi8, j = (wave << 6) + (a << 4) + laneN;
      c0r[a][r] = c0f[(size_t)(b0 + m) * HH + j];
      c1r[a][r] = c1f[(size_t)(b0 + m) * HH + j];
    }
  __syncthreads();

  for (int step = 0; step < PL; ++step) {
    if (tid < MT) A0[tid * 1056] = (__bf16)dinS[tid];
    // 1) hproj = h1 @ Wdec^T  (WMMA)
    {
      v8f ha[4];
#pragma unroll
      for (int g4 = 0; g4 < 4; ++g4)
#pragma unroll
        for (int r = 0; r < 8; ++r) ha[g4][r] = 0.f;
      gemm4_bf16<16>(A1 + 512, 1024, Wdec, HH, wave * 64, lane, ha);
#pragma unroll
      for (int g4 = 0; g4 < 4; ++g4)
#pragma unroll
        for (int r = 0; r < 8; ++r)
          hproj[(r + hi8) * HH + wave * 64 + g4 * 16 + laneN] = ha[g4][r];
    }
    __syncthreads();
    // 2) scores[m][s] = sum_h v[h]*tanh(EP[m][s][h] + hproj[m][h])
    {
      const int m = tid & 15;
      for (int k = 0; k < 32; ++k) {
        const int s = (tid >> 4) + (k << 4);
        const __bf16* pe = EP + ((size_t)(b0 + m) * SS + s) * HH;
        float a = 0.f;
        for (int h = 0; h < HH; h += 8) {
          union { uint4 u; __bf16 e[8]; } d;
          d.u = *(const uint4*)(pe + h);
#pragma unroll
          for (int qq = 0; qq < 8; ++qq)
            a += vv[h + qq] * tanhf((float)d.e[qq] + hproj[m * HH + h + qq]);
        }
        scw[m * HH + s] = a;
      }
    }
    __syncthreads();
    // 3) softmax over s, 2 rows per wave
    {
#pragma unroll 1
      for (int i = 0; i < 2; ++i) {
        const int m = wave * 2 + i;
        float vals[16], mx = -1e30f;
#pragma unroll
        for (int k = 0; k < 16; ++k) {
          vals[k] = scw[m * HH + lane + 32 * k];
          mx = fmaxf(mx, vals[k]);
        }
        for (int off = 16; off; off >>= 1) mx = fmaxf(mx, __shfl_xor(mx, off, 32));
        float sm = 0.f;
#pragma unroll
        for (int k = 0; k < 16; ++k) { vals[k] = __expf(vals[k] - mx); sm += vals[k]; }
        for (int off = 16; off; off >>= 1) sm += __shfl_xor(sm, off, 32);
        const float inv = 1.f / sm;
#pragma unroll
        for (int k = 0; k < 16; ++k) scw[m * HH + lane + 32 * k] = vals[k] * inv;
      }
    }
    __syncthreads();
    // 4) context[m][h] = sum_s w[m][s]*EO[m][s][h] -> A0 cols 1..512
    {
      const int m = tid >> 4, hb = (tid & 15) << 5;
      float ca[32];
#pragma unroll
      for (int qq = 0; qq < 32; ++qq) ca[qq] = 0.f;
      for (int s = 0; s < SS; ++s) {
        const float w = scw[m * HH + s];
        const __bf16* pe = EO + ((size_t)(b0 + m) * SS + s) * HH + hb;
#pragma unroll
        for (int q8 = 0; q8 < 32; q8 += 8) {
          union { uint4 u; __bf16 e[8]; } d;
          d.u = *(const uint4*)(pe + q8);
#pragma unroll
          for (int qq = 0; qq < 8; ++qq) ca[q8 + qq] += w * (float)d.e[qq];
        }
      }
#pragma unroll
      for (int qq = 0; qq < 32; ++qq) A0[m * 1056 + 1 + hb + qq] = (__bf16)ca[qq];
    }
    __syncthreads();
    // 5) LSTM d0 (K = 1 + 512 ctx + 512 h0 -> padded 1056)
    v8f acc[16];
#pragma unroll
    for (int j = 0; j < 16; ++j) {
      const float bv = bd0c[n0w + j * 16 + laneN];
#pragma unroll
      for (int r = 0; r < 8; ++r) acc[j][r] = bv;
    }
#pragma unroll 1
    for (int a = 0; a < 4; ++a)
      gemm4_bf16<33>(A0, 1056, Wd0, 1056, n0w + a * 64, lane, &acc[4 * a]);
    __syncthreads();
#pragma unroll
    for (int a = 0; a < 4; ++a)
      lstm_group(&acc[4 * a], c0r[a], wave, a, lane, A0 + 513, 1056, A1, 1024,
                 nullptr, 0);
    __syncthreads();
    // 6) LSTM d1
#pragma unroll
    for (int j = 0; j < 16; ++j) {
      const float bv = bd1c[n0w + j * 16 + laneN];
#pragma unroll
      for (int r = 0; r < 8; ++r) acc[j][r] = bv;
    }
#pragma unroll 1
    for (int a = 0; a < 4; ++a)
      gemm4_bf16<32>(A1, 1024, Wd1, 1024, n0w + a * 64, lane, &acc[4 * a]);
    __syncthreads();
#pragma unroll
    for (int a = 0; a < 4; ++a)
      lstm_group(&acc[4 * a], c1r[a], wave, a, lane, A1 + 512, 1024, nullptr, 0,
                 nullptr, 0);
    __syncthreads();
    // 7) pred + dec_in feedback
    if (tid < MT) {
      float p = fcb[0];
      for (int h = 0; h < HH; ++h)
        p += (float)A1[tid * 1024 + 512 + h] * fcw[h];
      out[(size_t)(b0 + tid) * PL + step] = p;
      dinS[tid] = 0.8f * p + 0.2f * histS[tid];
    }
    __syncthreads();
  }
}

// ------------------------------ launcher -----------------------------------
extern "C" void kernel_launch(void* const* d_in, const int* in_sizes, int n_in,
                              void* d_out, int out_size, void* d_ws, size_t ws_size,
                              hipStream_t stream) {
  (void)in_sizes; (void)n_in; (void)out_size; (void)ws_size;
  const float* src    = (const float*)d_in[0];
  const float* ew0    = (const float*)d_in[1];
  const float* eu0    = (const float*)d_in[2];
  const float* ebi0   = (const float*)d_in[3];
  const float* ebh0   = (const float*)d_in[4];
  const float* ew1    = (const float*)d_in[5];
  const float* eu1    = (const float*)d_in[6];
  const float* ebi1   = (const float*)d_in[7];
  const float* ebh1   = (const float*)d_in[8];
  const float* dw0    = (const float*)d_in[9];
  const float* du0    = (const float*)d_in[10];
  const float* dbi0   = (const float*)d_in[11];
  const float* dbh0   = (const float*)d_in[12];
  const float* dw1    = (const float*)d_in[13];
  const float* du1    = (const float*)d_in[14];
  const float* dbi1   = (const float*)d_in[15];
  const float* dbh1   = (const float*)d_in[16];
  const float* attn_W = (const float*)d_in[17];
  const float* attn_b = (const float*)d_in[18];
  const float* v_W    = (const float*)d_in[19];
  const float* fc_W   = (const float*)d_in[20];
  const float* fc_b   = (const float*)d_in[21];
  float* out = (float*)d_out;

  char* ws = (char*)d_ws;
  size_t off = 0;
  auto take = [&](size_t bytes) -> char* {
    char* p = ws + off;
    off = (off + bytes + 255) & ~(size_t)255;
    return p;
  };
  __bf16* Wc0  = (__bf16*)take(2048ull * 544 * 2);
  __bf16* Wc1  = (__bf16*)take(2048ull * 1024 * 2);
  __bf16* Wd0  = (__bf16*)take(2048ull * 1056 * 2);
  __bf16* Wd1  = (__bf16*)take(2048ull * 1024 * 2);
  __bf16* Wenc = (__bf16*)take(512ull * 512 * 2);
  __bf16* Wdec = (__bf16*)take(512ull * 512 * 2);
  float*  b0c  = (float*)take(2048ull * 4);
  float*  b1c  = (float*)take(2048ull * 4);
  float*  bd0c = (float*)take(2048ull * 4);
  float*  bd1c = (float*)take(2048ull * 4);
  __bf16* h0f  = (__bf16*)take((size_t)BB * HH * 2);
  __bf16* h1f  = (__bf16*)take((size_t)BB * HH * 2);
  float*  c0f  = (float*)take((size_t)BB * HH * 4);
  float*  c1f  = (float*)take((size_t)BB * HH * 4);
  __bf16* EO   = (__bf16*)take((size_t)BB * SS * HH * 2);   // enc_outs
  __bf16* EP   = (__bf16*)take((size_t)BB * SS * HH * 2);   // enc_proj

  k_convert_lstm<<<2048, 256, 0, stream>>>(ew0,  32, eu0, ebi0, ebh0, Wc0,  544, b0c);
  k_convert_lstm<<<2048, 256, 0, stream>>>(ew1, 512, eu1, ebi1, ebh1, Wc1, 1024, b1c);
  k_convert_lstm<<<2048, 256, 0, stream>>>(dw0, 513, du0, dbi0, dbh0, Wd0, 1056, bd0c);
  k_convert_lstm<<<2048, 256, 0, stream>>>(dw1, 512, du1, dbi1, dbh1, Wd1, 1024, bd1c);
  k_convert_attn<<<512, 256, 0, stream>>>(attn_W, Wenc, Wdec);

  const int encSmem = 2 * MT * 544 * 2 + 2 * MT * 1024 * 2;       // 100352 B
  k_encoder<<<BB / MT, 256, encSmem, stream>>>(src, Wc0, b0c, Wc1, b1c,
                                               EO, h0f, h1f, c0f, c1f);

  k_encproj<<<(BB * SS) / 32, 256, 0, stream>>>(EO, Wenc, attn_b, EP);

  const int decSmem = MT * 1056 * 2 + MT * 1024 * 2 +
                      (MT * HH) * 4 * 2 + HH * 4 * 2 + MT * 4 * 2;  // ~136 KB
  k_decoder<<<BB / MT, 256, decSmem, stream>>>(src, Wd0, bd0c, Wd1, bd1c, Wdec,
                                               EO, EP, h0f, h1f, c0f, c1f,
                                               v_W, fc_W, fc_b, out);
}
